// SambaYAttention_43980465111722
// MI455X (gfx1250) — matrix operand order repaired and checked by reading, not compiled, not run on hardware
//
#include <hip/hip_runtime.h>
#include <hip/hip_bf16.h>

// ---------------- types ----------------
typedef __attribute__((ext_vector_type(16))) __bf16 v16bf;
typedef __attribute__((ext_vector_type(8)))  __bf16 v8bf;
typedef __attribute__((ext_vector_type(4)))  __bf16 v4bf;
typedef __attribute__((ext_vector_type(8)))  float  v8f;

#define H_    16
#define KV_   8
#define D_    128
#define HID_  2048
#define B_    2
#define S_    2048
#define QKVN  4096
#define LAMBDA_INIT_ 0.35550906759096924f
#define EPS_  1e-5f
#define SCALE_ 0.08838834764831845f   // 128^-0.5

// ---------------- CDNA5 async copy helpers ----------------
// global -> LDS direct copy (no VGPR round trip), tracked by ASYNCcnt.
static __device__ __forceinline__ void async_ld_b128(void* lds, const void* gmem) {
  unsigned la = (unsigned)(unsigned long long)lds;   // low 32 bits = DS offset
  asm volatile("global_load_async_to_lds_b128 %0, %1, off"
               :: "v"(la), "v"(gmem) : "memory");
}
static __device__ __forceinline__ void wait_async0() {
  asm volatile("s_wait_asynccnt 0x0" ::: "memory");
}

// ---------------- WMMA helpers ----------------
static __device__ __forceinline__ v8f wmma_bf16(v16bf a, v16bf b, v8f c) {
  return __builtin_amdgcn_wmma_f32_16x16x32_bf16(false, a, false, b, (short)0, c, false, false);
}

// A-matrix 16x32 bf16 fragment from row-major source (lane m = lane&15,
// K contiguous 16 values starting at h16, matching ldfragB).
static __device__ __forceinline__ v16bf ldfragA(const __bf16* base, int ld) {
  const int lane = threadIdx.x & 31;
  const __bf16* p = base + (lane & 15) * ld + ((lane >> 4) << 4);
  v8bf lo = *(const v8bf*)p;
  v8bf hi = *(const v8bf*)(p + 8);
  return __builtin_shufflevector(lo, hi, 0,1,2,3,4,5,6,7,8,9,10,11,12,13,14,15);
}

// B-matrix 32x16 bf16 fragment: lane n = lane&15 holds column n, K contiguous
// 16 values starting at h16 (0 or 16) -> one 32-byte contiguous load.
static __device__ __forceinline__ v16bf ldfragB(const __bf16* base, int ld) {
  const int lane = threadIdx.x & 31;
  const __bf16* p = base + (lane & 15) * ld + ((lane >> 4) << 4);
  return *(const v16bf*)p;
}

// ---------------- f32 -> bf16 conversion ----------------
__global__ __launch_bounds__(256) void cvt_f32_bf16(const float* __restrict__ in,
                                                    __bf16* __restrict__ out, int n4) {
  int i = blockIdx.x * 256 + threadIdx.x;
  if (i < n4) {
    float4 f = ((const float4*)in)[i];
    v4bf v;
    v[0] = (__bf16)f.x; v[1] = (__bf16)f.y; v[2] = (__bf16)f.z; v[3] = (__bf16)f.w;
    ((v4bf*)out)[i] = v;
  }
}

// ---------------- lambda scalar ----------------
__global__ __launch_bounds__(128) void lambda_kernel(const float* __restrict__ lq1,
                                                     const float* __restrict__ lk1,
                                                     const float* __restrict__ lq2,
                                                     const float* __restrict__ lk2,
                                                     float* __restrict__ lam_out) {
  __shared__ float r1[128], r2[128];
  int t = threadIdx.x;
  r1[t] = lq1[t] * lk1[t];
  r2[t] = lq2[t] * lk2[t];
  __syncthreads();
  for (int s = 64; s > 0; s >>= 1) {
    if (t < s) { r1[t] += r1[t + s]; r2[t] += r2[t + s]; }
    __syncthreads();
  }
  if (t == 0) *lam_out = __expf(r1[0]) - __expf(r2[0]) + LAMBDA_INIT_;
}

// ---------------- bf16 WMMA GEMM: C[M,N] = A[M,K] @ Bw[N,K]^T + bias ----------------
// block tile 128x128, 256 threads = 8 waves in 4(M) x 2(N), wave tile 32x64.
// Double-buffered LDS, filled via ASYNCcnt-tracked global_load_async_to_lds_b128.
__global__ __launch_bounds__(256) void gemm_bf16(const __bf16* __restrict__ A,
                                                 const __bf16* __restrict__ Bw,
                                                 const float* __restrict__ bias,
                                                 void* __restrict__ Cout,
                                                 int M, int N, int K, int bf16out) {
  __shared__ __attribute__((aligned(64))) __bf16 As[2][128 * 32];
  __shared__ __attribute__((aligned(64))) __bf16 Bs[2][128 * 32];
  const int tid = threadIdx.x, lane = tid & 31, wave = tid >> 5;
  const int wm = wave & 3, wn = wave >> 2;
  const size_t bm = (size_t)blockIdx.y * 128, bn = (size_t)blockIdx.x * 128;

  v8f acc[2][4];
  v8f zz = {};
  #pragma unroll
  for (int i = 0; i < 2; i++)
    #pragma unroll
    for (int j = 0; j < 4; j++) acc[i][j] = zz;

  // async-stage one 128x32 K-slab of A and B into LDS buffer `buf`
  auto stage = [&](int buf, int k0) {
    #pragma unroll
    for (int i = tid; i < 512; i += 256) {
      int r = i >> 2, c = (i & 3) << 3;
      async_ld_b128(&As[buf][r * 32 + c], A  + (bm + r) * (size_t)K + k0 + c);
      async_ld_b128(&Bs[buf][r * 32 + c], Bw + (bn + r) * (size_t)K + k0 + c);
    }
  };

  stage(0, 0);
  wait_async0();
  __syncthreads();

  int cur = 0;
  for (int k0 = 0; k0 < K; k0 += 32) {
    if (k0 + 32 < K) stage(cur ^ 1, k0 + 32);   // prefetch next slab (overlaps WMMA)

    v16bf af[2], bf[4];
    #pragma unroll
    for (int mi = 0; mi < 2; mi++) af[mi] = ldfragA(&As[cur][(wm * 32 + mi * 16) * 32], 32);
    #pragma unroll
    for (int ni = 0; ni < 4; ni++) bf[ni] = ldfragB(&Bs[cur][(wn * 64 + ni * 16) * 32], 32);
    #pragma unroll
    for (int mi = 0; mi < 2; mi++)
      #pragma unroll
      for (int ni = 0; ni < 4; ni++)
        acc[mi][ni] = wmma_bf16(af[mi], bf[ni], acc[mi][ni]);

    wait_async0();        // next slab resident
    __syncthreads();      // all waves done reading `cur`, next slab visible
    cur ^= 1;
  }

  const int ln = lane & 15, hi8 = (lane >> 4) << 3;
  #pragma unroll
  for (int mi = 0; mi < 2; mi++)
    #pragma unroll
    for (int ni = 0; ni < 4; ni++) {
      size_t col = bn + wn * 64 + ni * 16 + ln;
      float bv = bias[col];
      #pragma unroll
      for (int r = 0; r < 8; r++) {
        size_t row = bm + wm * 32 + mi * 16 + r + hi8;
        float v = acc[mi][ni][r] + bv;
        if (bf16out) ((__bf16*)Cout)[row * (size_t)N + col] = (__bf16)v;
        else         ((float*)Cout)[row * (size_t)N + col] = v;
      }
    }
}

// ---------------- differential flash attention ----------------
// grid: (S/64, B*8). block 256 = 8 waves. waves 0-3: map1, waves 4-7: map2.
// Each wave: 16 query rows, online softmax over key blocks of 64, O[16x256] f32.
__global__ __launch_bounds__(256) void diffattn(const __bf16* __restrict__ qkv,
                                                const float* __restrict__ lam_p,
                                                const float* __restrict__ subln_w,
                                                __bf16* __restrict__ attn_out) {
  extern __shared__ char smem[];
  __bf16* kbuf = (__bf16*)smem;                     // [64][256] (k1|k2 halves)
  __bf16* vtb  = (__bf16*)(smem + 64 * 256 * 2);    // [256][64] transposed V
  __bf16* pst  = (__bf16*)(smem + 2 * 64 * 256 * 2);// [8 waves][16][64] P staging
  float*  o2b  = (float*)smem;                      // [64][256] aliases kbuf+vtb (post-loop)

  const int qt = blockIdx.x;
  const int b  = blockIdx.y >> 3;
  const int hp = blockIdx.y & 7;
  const int tid = threadIdx.x, lane = tid & 31, wave = tid >> 5;
  const int map = wave >> 2, wsub = wave & 3;
  const int ln = lane & 15, hi8 = (lane >> 4) << 3;

  const int kp = hp >> 1;                    // GQA: kv pair index
  const size_t row0 = (size_t)b * S_;
  const int qcol = (2 * hp + map) * D_;
  const int kcol = H_ * D_ + 2 * kp * D_;    // k1|k2 (256 cols)
  const int vcol = H_ * D_ + KV_ * D_ + 2 * kp * D_;
  const int qrow0 = qt * 64 + wsub * 16;

  // resident Q fragments (K = 128 -> 4 frags)
  v16bf qf[4];
  #pragma unroll
  for (int kk = 0; kk < 4; kk++)
    qf[kk] = ldfragA(qkv + (row0 + qrow0) * QKVN + qcol + kk * 32, QKVN);

  v8f o[16];
  v8f zz = {};
  #pragma unroll
  for (int t = 0; t < 16; t++) o[t] = zz;
  float mi_[8], li[8];
  #pragma unroll
  for (int r = 0; r < 8; r++) { mi_[r] = -1e30f; li[r] = 0.f; }

  const float lam = lam_p[0];

  for (int kb = 0; kb <= qt; ++kb) {
    __syncthreads();
    // K tile [64][256]: direct global->LDS async copy (ASYNCcnt)
    for (int i = tid; i < 2048; i += 256) {
      int r = i >> 5, c = (i & 31) << 3;
      async_ld_b128(kbuf + r * 256 + c,
                    qkv + (row0 + kb * 64 + r) * QKVN + kcol + c);
    }
    // V tile transposed [256][64]: VGPR path (needs transpose)
    for (int i = tid; i < 2048; i += 256) {
      int r = i >> 5, c = (i & 31) << 3;
      v8bf vv = *(const v8bf*)(qkv + (row0 + kb * 64 + r) * QKVN + vcol + c);
      #pragma unroll
      for (int e = 0; e < 8; e++) vtb[(c + e) * 64 + r] = vv[e];
    }
    wait_async0();
    __syncthreads();

    // scores S = q @ k^T   (16x64, 4 n-tiles x 4 k-steps = 16 WMMAs)
    v8f s[4];
    #pragma unroll
    for (int nt = 0; nt < 4; nt++) s[nt] = zz;
    const __bf16* kb_m = kbuf + map * D_;
    #pragma unroll
    for (int nt = 0; nt < 4; nt++)
      #pragma unroll
      for (int kk = 0; kk < 4; kk++)
        s[nt] = wmma_bf16(qf[kk], ldfragB(kb_m + (nt * 16) * 256 + kk * 32, 256), s[nt]);

    // scale + causal mask (only diagonal block needs masking)
    const bool diag = (kb == qt);
    #pragma unroll
    for (int nt = 0; nt < 4; nt++)
      #pragma unroll
      for (int r = 0; r < 8; r++) {
        float v = s[nt][r] * SCALE_;
        if (diag) {
          int qg = qrow0 + r + hi8;
          int kg = qt * 64 + nt * 16 + ln;
          if (kg > qg) v = -1e30f;
        }
        s[nt][r] = v;
      }

    // online softmax; row stats live per-16-lane-half (rows r and r+8)
    __bf16* pw = pst + wave * (16 * 64);
    #pragma unroll
    for (int r = 0; r < 8; r++) {
      float mx = fmaxf(fmaxf(s[0][r], s[1][r]), fmaxf(s[2][r], s[3][r]));
      mx = fmaxf(mx, __shfl_xor(mx, 1, 32));
      mx = fmaxf(mx, __shfl_xor(mx, 2, 32));
      mx = fmaxf(mx, __shfl_xor(mx, 4, 32));
      mx = fmaxf(mx, __shfl_xor(mx, 8, 32));
      float mnew  = fmaxf(mi_[r], mx);
      float alpha = __expf(mi_[r] - mnew);
      mi_[r] = mnew;
      float rs = 0.f;
      #pragma unroll
      for (int nt = 0; nt < 4; nt++) {
        float p = __expf(s[nt][r] - mnew);
        s[nt][r] = p;
        rs += p;
      }
      rs += __shfl_xor(rs, 1, 32);
      rs += __shfl_xor(rs, 2, 32);
      rs += __shfl_xor(rs, 4, 32);
      rs += __shfl_xor(rs, 8, 32);
      li[r] = li[r] * alpha + rs;
      #pragma unroll
      for (int t = 0; t < 16; t++) o[t][r] *= alpha;
      // C-layout -> row-major bf16 P into per-wave LDS staging
      #pragma unroll
      for (int nt = 0; nt < 4; nt++)
        pw[(r + hi8) * 64 + nt * 16 + ln] = (__bf16)s[nt][r];
    }
    asm volatile("s_wait_dscnt 0x0" ::: "memory");  // P stores visible to own reads

    // O += P @ V    (2 k-steps x 16 c-tiles = 32 WMMAs)
    #pragma unroll
    for (int kk = 0; kk < 2; kk++) {
      v16bf pf = ldfragA(pw + kk * 32, 64);
      #pragma unroll
      for (int ct = 0; ct < 16; ct++)
        o[ct] = wmma_bf16(pf, ldfragB(vtb + (ct * 16) * 64 + kk * 32, 64), o[ct]);
    }
  }

  // normalize by softmax denominator
  #pragma unroll
  for (int r = 0; r < 8; r++) {
    float inv = 1.f / li[r];
    #pragma unroll
    for (int t = 0; t < 16; t++) o[t][r] *= inv;
  }

  __syncthreads();
  if (map == 1) {  // publish lam * O2
    #pragma unroll
    for (int t = 0; t < 16; t++)
      #pragma unroll
      for (int r = 0; r < 8; r++)
        o2b[(wsub * 16 + r + hi8) * 256 + t * 16 + ln] = o[t][r] * lam;
  }
  __syncthreads();
  if (map == 0) {  // combine + RMSNorm + subln epilogue
    #pragma unroll
    for (int r = 0; r < 8; r++) {
      float ss = 0.f;
      #pragma unroll
      for (int t = 0; t < 16; t++) {
        float v = o[t][r] - o2b[(wsub * 16 + r + hi8) * 256 + t * 16 + ln];
        o[t][r] = v;
        ss += v * v;
      }
      ss += __shfl_xor(ss, 1, 32);
      ss += __shfl_xor(ss, 2, 32);
      ss += __shfl_xor(ss, 4, 32);
      ss += __shfl_xor(ss, 8, 32);
      float rms = rsqrtf(ss * (1.f / 256.f) + EPS_);
      size_t orow = row0 + qrow0 + r + hi8;
      #pragma unroll
      for (int t = 0; t < 16; t++) {
        int c = t * 16 + ln;
        float v = o[t][r] * rms * subln_w[c] * (1.f - LAMBDA_INIT_);
        attn_out[orow * (size_t)(H_ * D_) + hp * 256 + c] = (__bf16)v;
      }
    }
  }
}

// ---------------- launch ----------------
extern "C" void kernel_launch(void* const* d_in, const int* in_sizes, int n_in,
                              void* d_out, int out_size, void* d_ws, size_t ws_size,
                              hipStream_t stream) {
  const float* hs   = (const float*)d_in[0];
  const float* wqkv = (const float*)d_in[1];
  const float* bqkv = (const float*)d_in[2];
  const float* lq1  = (const float*)d_in[3];
  const float* lk1  = (const float*)d_in[4];
  const float* lq2  = (const float*)d_in[5];
  const float* lk2  = (const float*)d_in[6];
  const float* slnw = (const float*)d_in[7];
  const float* outw = (const float*)d_in[8];
  const float* outb = (const float*)d_in[9];
  float* out = (float*)d_out;
  (void)in_sizes; (void)n_in; (void)out_size; (void)ws_size;

  char* ws = (char*)d_ws;
  float*  lam  = (float*)ws;                              // 256 B slot
  __bf16* hsb  = (__bf16*)(ws + 256);                     // [4096][2048]
  __bf16* wqb  = hsb  + (size_t)4096 * 2048;              // [4096][2048]
  __bf16* qkvb = wqb  + (size_t)4096 * 2048;              // [4096][4096]
  __bf16* attb = qkvb + (size_t)4096 * 4096;              // [4096][2048]
  __bf16* owb  = attb + (size_t)4096 * 2048;              // [2048][2048]

  const int n4_big = 4096 * 2048 / 4;   // hidden / Wqkv
  const int n4_ow  = 2048 * 2048 / 4;   // out_w
  cvt_f32_bf16<<<(n4_big + 255) / 256, 256, 0, stream>>>(hs,   hsb, n4_big);
  cvt_f32_bf16<<<(n4_big + 255) / 256, 256, 0, stream>>>(wqkv, wqb, n4_big);
  cvt_f32_bf16<<<(n4_ow  + 255) / 256, 256, 0, stream>>>(outw, owb, n4_ow);
  lambda_kernel<<<1, 128, 0, stream>>>(lq1, lk1, lq2, lk2, lam);

  // qkv = hidden @ Wqkv^T + b : M=4096 N=4096 K=2048, bf16 out
  gemm_bf16<<<dim3(32, 32), 256, 0, stream>>>(hsb, wqb, bqkv, qkvb, 4096, 4096, 2048, 1);

  // differential attention: grid (S/64, B*8), 80 KB dynamic LDS (fits 320 KB/WGP)
  diffattn<<<dim3(32, 16), 256, 81920, stream>>>(qkvb, lam, slnw, attb);

  // out = attn @ out_w^T + b : M=4096 N=2048 K=2048, f32 out
  gemm_bf16<<<dim3(16, 32), 256, 0, stream>>>(attb, owb, outb, out, 4096, 2048, 2048, 0);
}
